// MultiheadAttention_63745904607697
// MI455X (gfx1250) — compile-verified, matrix operand order checked by reading, not run on hardware
//
#include <hip/hip_runtime.h>
#include <hip/hip_bf16.h>

#define D_MODEL   1024
#define NUM_HEADS 16
#define HEAD_DIM  64
#define BATCH     512
#define SEQ       64
#define MTOK      (BATCH * SEQ)   // 32768 tokens

typedef __attribute__((ext_vector_type(16))) __bf16         v16bf;
typedef __attribute__((ext_vector_type(8)))  float          v8f;
typedef __attribute__((ext_vector_type(4)))  float          v4f;
typedef __attribute__((ext_vector_type(8)))  unsigned short v8us;
typedef __attribute__((ext_vector_type(4)))  int            v4i;

#define AS_GLOBAL __attribute__((address_space(1)))
#define AS_LDS    __attribute__((address_space(3)))

// ---------------------------------------------------------------------------
// helpers
// ---------------------------------------------------------------------------
__device__ __forceinline__ unsigned short f32_to_bf16(float f) {
  unsigned int x = __float_as_uint(f);
  x += 0x7FFFu + ((x >> 16) & 1u);        // round-to-nearest-even
  return (unsigned short)(x >> 16);
}

// Load a 16x32 bf16 WMMA A/B fragment from K-major (row-major MxK / NxK)
// memory. Matches the CDNA5 16-bit A/B VGPR layout:
//   lane L (0..15):  row L,    halves 0..7 = K 0..7,  halves 8..15 = K 16..23
//   lane L (16..31): row L-16, halves 0..7 = K 8..15, halves 8..15 = K 24..31
// => two {global,ds}_load_b128 per lane, fully coalesced.
__device__ __forceinline__ v16bf load_frag(const unsigned short* base, int ld, int lane) {
  const int r  = lane & 15;
  const int hi = lane >> 4;
  const unsigned short* p = base + (size_t)r * ld + hi * 8;
  union { v8us u[2]; v16bf f; } u;
  u.u[0] = *(const v8us*)(p);
  u.u[1] = *(const v8us*)(p + 16);
  return u.f;
}

__device__ __forceinline__ v8f wmma_bf16(v16bf a, v16bf b, v8f c) {
  return __builtin_amdgcn_wmma_f32_16x16x32_bf16(
      /*neg_a=*/false, a, /*neg_b=*/false, b,
      /*c_mod=*/(short)0, c, /*reuse_a=*/false, /*reuse_b=*/false);
}

// ---------------------------------------------------------------------------
// stage 1: fp32 -> bf16 conversion (8 elems / thread, b128 in & out)
// ---------------------------------------------------------------------------
__global__ __launch_bounds__(256) void cvt_bf16_kernel(const float* __restrict__ in,
                                                       unsigned short* __restrict__ out,
                                                       int n8) {   // n/8 chunks
  int i = blockIdx.x * 256 + threadIdx.x;
  if (i >= n8) return;
  v4f lo = ((const v4f*)in)[2 * i];
  v4f hi = ((const v4f*)in)[2 * i + 1];
  v8us o;
#pragma unroll
  for (int k = 0; k < 4; ++k) { o[k] = f32_to_bf16(lo[k]); o[k + 4] = f32_to_bf16(hi[k]); }
  ((v8us*)out)[i] = o;
}

// ---------------------------------------------------------------------------
// stage 2/4: GEMM  C[M,N] = A[M,K] * B[N,K]^T + bias  (both operands K-major)
// block = 4 waves; block tile 128(M) x 64(N); wave tile 32(M) x 64(N).
// Register double-buffered k-pipeline: next k-step fragments are in flight
// while the current k-step's 8 WMMAs execute.
// mode 0: fp32 store to out[m*N+n]                      (output projection)
// mode 1: bias + RoPE, bf16 store to (b,h,s,d)          (Q and K)
// mode 2: bias,        bf16 store to (b,h,d,s)          (V transposed)
// ---------------------------------------------------------------------------
__global__ __launch_bounds__(128) void gemm_bf16_nt_kernel(
    const unsigned short* __restrict__ A,
    const unsigned short* __restrict__ B,
    const float* __restrict__ bias,
    void* __restrict__ out,
    int M, int N, int K, int mode) {
  const int lane  = threadIdx.x & 31;
  const int wave  = threadIdx.x >> 5;
  const int n_blk = blockIdx.x * 64;
  const int m0    = blockIdx.y * 128 + wave * 32;   // wave owns rows [m0, m0+32)

  v8f acc[2][4] = {};

  const unsigned short* A0 = A + (size_t)m0 * K;
  const unsigned short* A1 = A + (size_t)(m0 + 16) * K;
  const unsigned short* Bp = B + (size_t)n_blk * K;

  v16bf a_cur[2], b_cur[4];
  a_cur[0] = load_frag(A0, K, lane);
  a_cur[1] = load_frag(A1, K, lane);
#pragma unroll
  for (int j = 0; j < 4; ++j) b_cur[j] = load_frag(Bp + (size_t)j * 16 * K, K, lane);

#pragma unroll 2
  for (int k0 = 0; k0 < K - 32; k0 += 32) {
    const int k1 = k0 + 32;
    // issue next k-step loads first -> overlap with WMMAs below
    v16bf a_nxt[2], b_nxt[4];
    a_nxt[0] = load_frag(A0 + k1, K, lane);
    a_nxt[1] = load_frag(A1 + k1, K, lane);
#pragma unroll
    for (int j = 0; j < 4; ++j) b_nxt[j] = load_frag(Bp + (size_t)j * 16 * K + k1, K, lane);

#pragma unroll
    for (int j = 0; j < 4; ++j) {
      acc[0][j] = wmma_bf16(a_cur[0], b_cur[j], acc[0][j]);
      acc[1][j] = wmma_bf16(a_cur[1], b_cur[j], acc[1][j]);
    }
    a_cur[0] = a_nxt[0];
    a_cur[1] = a_nxt[1];
#pragma unroll
    for (int j = 0; j < 4; ++j) b_cur[j] = b_nxt[j];
  }
  // final k-step
#pragma unroll
  for (int j = 0; j < 4; ++j) {
    acc[0][j] = wmma_bf16(a_cur[0], b_cur[j], acc[0][j]);
    acc[1][j] = wmma_bf16(a_cur[1], b_cur[j], acc[1][j]);
  }

  // C layout: element v of lane L is (m = m0 + v + (L>>4)*8, n = n_blk + j*16 + (L&15))
  const int r  = lane & 15;
  const int hi = lane >> 4;
#pragma unroll
  for (int mi = 0; mi < 2; ++mi) {
    const int mbase = m0 + mi * 16;
#pragma unroll
    for (int j = 0; j < 4; ++j) {
      const int n  = n_blk + j * 16 + r;
      const float bn = bias[n];
      if (mode == 0) {
        float* o = (float*)out;
#pragma unroll
        for (int v = 0; v < 8; ++v) {
          const int m = mbase + v + hi * 8;
          o[(size_t)m * N + n] = acc[mi][j][v] + bn;
        }
      } else if (mode == 1) {
        unsigned short* o = (unsigned short*)out;
        const int   pair = n >> 1;                                   // global pair index
        const float freq = __expf(-(float)pair * (9.210340371976184f / 512.0f)); // 10000^(-pair/512)
#pragma unroll
        for (int v = 0; v < 8; ++v) {
          const int   m = mbase + v + hi * 8;
          const float c = acc[mi][j][v] + bn;
          const float p = __shfl_xor(c, 1, 32);   // partner channel (n^1) lives in lane^1
          const int   s = m & (SEQ - 1);
          float sn, cs;
          __sincosf((float)s * freq, &sn, &cs);
          const float ro = (n & 1) ? (p * sn + c * cs)    // odd:  xe*sin + xo*cos
                                   : (c * cs - p * sn);   // even: xe*cos - xo*sin
          const int b_ = m >> 6, h_ = n >> 6, d_ = n & 63;
          o[(((size_t)(b_ * NUM_HEADS + h_) * SEQ) + s) * HEAD_DIM + d_] = f32_to_bf16(ro);
        }
      } else {  // mode 2: V, stored transposed (b,h,d,s)
        unsigned short* o = (unsigned short*)out;
#pragma unroll
        for (int v = 0; v < 8; ++v) {
          const int   m = mbase + v + hi * 8;
          const float c = acc[mi][j][v] + bn;
          const int b_ = m >> 6, s = m & 63, h_ = n >> 6, d_ = n & 63;
          o[(((size_t)(b_ * NUM_HEADS + h_) * HEAD_DIM) + d_) * SEQ + s] = f32_to_bf16(c);
        }
      }
    }
  }
}

// ---------------------------------------------------------------------------
// stage 3: attention for one (batch, head). 4 waves; wave w owns query rows
// [16w, 16w+16). Q,K: (s,d) bf16; Vt: (d,s) bf16; O: token-major (b,s,h,d) bf16.
// Staging uses the CDNA5 direct global->LDS async path when available.
// ---------------------------------------------------------------------------
__global__ __launch_bounds__(128) void attn_kernel(
    const unsigned short* __restrict__ Q,
    const unsigned short* __restrict__ K,
    const unsigned short* __restrict__ Vt,
    unsigned short* __restrict__ O) {
  __shared__ alignas(16) unsigned short sQ[SEQ * HEAD_DIM];   // (s,d)
  __shared__ alignas(16) unsigned short sK[SEQ * HEAD_DIM];   // (s,d)
  __shared__ alignas(16) unsigned short sV[HEAD_DIM * SEQ];   // (d,s)
  __shared__ alignas(16) float          sS[SEQ * SEQ];        // scores fp32
  __shared__ alignas(16) unsigned short sP[SEQ * SEQ];        // probs bf16 (s_q, s_k)

  const int bh = blockIdx.x;                 // b*NUM_HEADS + h
  const unsigned short* gQ = Q  + (size_t)bh * SEQ * HEAD_DIM;
  const unsigned short* gK = K  + (size_t)bh * SEQ * HEAD_DIM;
  const unsigned short* gV = Vt + (size_t)bh * SEQ * HEAD_DIM;

#if __has_builtin(__builtin_amdgcn_global_load_async_to_lds_b128)
  // CDNA5 path: GLOBAL_LOAD_ASYNC_TO_LDS_B128, tracked by ASYNCcnt; no VGPR
  // round-trip for the staging copies. Param 1 is a non-const AS(1) int4*;
  // the uintptr_t round-trip is numerically exact for global pointers.
  // Param 2 (LDS) uses a real addrspacecast (generic->AS3 translation).
  for (int i = threadIdx.x; i < (SEQ * HEAD_DIM) / 8; i += 128) {
    __builtin_amdgcn_global_load_async_to_lds_b128(
        (AS_GLOBAL v4i*)(uintptr_t)(gQ + (size_t)i * 8),
        (AS_LDS v4i*)(sQ + (size_t)i * 8), 0, 0);
    __builtin_amdgcn_global_load_async_to_lds_b128(
        (AS_GLOBAL v4i*)(uintptr_t)(gK + (size_t)i * 8),
        (AS_LDS v4i*)(sK + (size_t)i * 8), 0, 0);
    __builtin_amdgcn_global_load_async_to_lds_b128(
        (AS_GLOBAL v4i*)(uintptr_t)(gV + (size_t)i * 8),
        (AS_LDS v4i*)(sV + (size_t)i * 8), 0, 0);
  }
#if __has_builtin(__builtin_amdgcn_s_wait_asynccnt)
  __builtin_amdgcn_s_wait_asynccnt(0);
#else
  asm volatile("s_wait_asynccnt 0x0" ::: "memory");
#endif
#else
  // fallback: staged b128 copy through VGPRs
  for (int i = threadIdx.x; i < (SEQ * HEAD_DIM) / 8; i += 128) {
    ((v8us*)sQ)[i] = ((const v8us*)gQ)[i];
    ((v8us*)sK)[i] = ((const v8us*)gK)[i];
    ((v8us*)sV)[i] = ((const v8us*)gV)[i];
  }
#endif
  __syncthreads();

  const int lane = threadIdx.x & 31;
  const int wave = threadIdx.x >> 5;
  const int r    = lane & 15;
  const int hi   = lane >> 4;

  // S = (Q K^T) * 1/sqrt(64)
#pragma unroll
  for (int j = 0; j < 4; ++j) {
    v8f acc = {};
#pragma unroll
    for (int k0 = 0; k0 < HEAD_DIM; k0 += 32) {
      v16bf a = load_frag(sQ + wave * 16 * HEAD_DIM + k0, HEAD_DIM, lane);
      v16bf b = load_frag(sK + j * 16 * HEAD_DIM + k0, HEAD_DIM, lane);
      acc = wmma_bf16(a, b, acc);
    }
#pragma unroll
    for (int v = 0; v < 8; ++v)
      sS[(wave * 16 + v + hi * 8) * SEQ + j * 16 + r] = acc[v] * 0.125f;
  }
  __syncthreads();

  // row softmax -> bf16 probs (threads 0..63 each own one query row)
  if (threadIdx.x < SEQ) {
    float* row = sS + threadIdx.x * SEQ;
    float mx = -1e30f;
#pragma unroll 8
    for (int i = 0; i < SEQ; ++i) mx = fmaxf(mx, row[i]);
    float sum = 0.0f;
#pragma unroll 8
    for (int i = 0; i < SEQ; ++i) { float e = __expf(row[i] - mx); row[i] = e; sum += e; }
    const float inv = 1.0f / sum;
#pragma unroll 8
    for (int i = 0; i < SEQ; ++i)
      sP[threadIdx.x * SEQ + i] = f32_to_bf16(row[i] * inv);
  }
  __syncthreads();

  // O = P * V   (A = probs rows, B = V^T so K-dim = key position is contiguous)
  const int b_ = bh >> 4;
  const int h_ = bh & 15;
#pragma unroll
  for (int j = 0; j < 4; ++j) {
    v8f acc = {};
#pragma unroll
    for (int k0 = 0; k0 < SEQ; k0 += 32) {
      v16bf a = load_frag(sP + wave * 16 * SEQ + k0, SEQ, lane);
      v16bf b = load_frag(sV + j * 16 * SEQ + k0, SEQ, lane);
      acc = wmma_bf16(a, b, acc);
    }
#pragma unroll
    for (int v = 0; v < 8; ++v) {
      const int sq = wave * 16 + v + hi * 8;
      const int d_ = j * 16 + r;
      O[((size_t)(b_ * SEQ + sq)) * D_MODEL + h_ * HEAD_DIM + d_] = f32_to_bf16(acc[v]);
    }
  }
}

// ---------------------------------------------------------------------------
// launch
// ---------------------------------------------------------------------------
extern "C" void kernel_launch(void* const* d_in, const int* in_sizes, int n_in,
                              void* d_out, int out_size, void* d_ws, size_t ws_size,
                              hipStream_t stream) {
  const float* x  = (const float*)d_in[0];
  const float* Wq = (const float*)d_in[1];
  const float* bq = (const float*)d_in[2];
  const float* Wk = (const float*)d_in[3];
  const float* bk = (const float*)d_in[4];
  const float* Wv = (const float*)d_in[5];
  const float* bv = (const float*)d_in[6];
  const float* Wo = (const float*)d_in[7];
  const float* bo = (const float*)d_in[8];

  char* ws = (char*)d_ws;
  const size_t SZ_X = (size_t)MTOK * D_MODEL * sizeof(unsigned short);    // 64 MB
  const size_t SZ_W = (size_t)D_MODEL * D_MODEL * sizeof(unsigned short); //  2 MB

  unsigned short* Xbf = (unsigned short*)(ws);                 // X (bf16); reused later as O
  unsigned short* WqB = (unsigned short*)(ws + SZ_X);
  unsigned short* WkB = (unsigned short*)(ws + SZ_X + 1 * SZ_W);
  unsigned short* WvB = (unsigned short*)(ws + SZ_X + 2 * SZ_W);
  unsigned short* WoB = (unsigned short*)(ws + SZ_X + 3 * SZ_W);
  unsigned short* Qh  = (unsigned short*)(ws + SZ_X + 4 * SZ_W);          // (b,h,s,d) 64 MB
  unsigned short* Kh  = Qh + (size_t)MTOK * D_MODEL;                      // (b,h,s,d) 64 MB
  unsigned short* Vt  = Kh + (size_t)MTOK * D_MODEL;                      // (b,h,d,s) 64 MB

  // stage 1: conversions (8 elements per thread)
  const int nx8 = (MTOK * D_MODEL) / 8;
  cvt_bf16_kernel<<<(nx8 + 255) / 256, 256, 0, stream>>>(x, Xbf, nx8);
  const int nw8 = (D_MODEL * D_MODEL) / 8;
  cvt_bf16_kernel<<<(nw8 + 255) / 256, 256, 0, stream>>>(Wq, WqB, nw8);
  cvt_bf16_kernel<<<(nw8 + 255) / 256, 256, 0, stream>>>(Wk, WkB, nw8);
  cvt_bf16_kernel<<<(nw8 + 255) / 256, 256, 0, stream>>>(Wv, WvB, nw8);
  cvt_bf16_kernel<<<(nw8 + 255) / 256, 256, 0, stream>>>(Wo, WoB, nw8);

  // stage 2: Q/K/V projections (+bias, +RoPE for Q/K, transpose for V)
  dim3 gg(D_MODEL / 64, MTOK / 128);
  gemm_bf16_nt_kernel<<<gg, 128, 0, stream>>>(Xbf, WqB, bq, Qh, MTOK, D_MODEL, D_MODEL, 1);
  gemm_bf16_nt_kernel<<<gg, 128, 0, stream>>>(Xbf, WkB, bk, Kh, MTOK, D_MODEL, D_MODEL, 1);
  gemm_bf16_nt_kernel<<<gg, 128, 0, stream>>>(Xbf, WvB, bv, Vt, MTOK, D_MODEL, D_MODEL, 2);

  // stage 3: attention; O (bf16, token-major) overwrites Xbf (X is dead now)
  attn_kernel<<<BATCH * NUM_HEADS, 128, 0, stream>>>(Qh, Kh, Vt, Xbf);

  // stage 4: output projection, fp32 result
  gemm_bf16_nt_kernel<<<gg, 128, 0, stream>>>(Xbf, WoB, bo, d_out, MTOK, D_MODEL, D_MODEL, 0);
}
